// RNN_90409061580807
// MI455X (gfx1250) — matrix operand order repaired and checked by reading, not compiled, not run on hardware
//
#include <hip/hip_runtime.h>
#include <hip/hip_bf16.h>

// RNN LM for MI455X (gfx1250): bf16 WMMA (v_wmma_f32_16x16x32_bf16) everywhere.
// embed+cast -> big Xproj GEMM -> 2*T small step kernels -> big logits GEMM.
// 192MB L2 holds all GEMM operands; DRAM traffic ~= 328MB logits store.
// Register-pressure discipline: <=176 VGPRs live state per wave (no spills).
// Wout padded to 10240 rows so the logits GEMM uses <2,4> tiles with a
// store-only edge guard. Wave->tile mapping puts M-groups fastest so the 8
// waves of a block share the larger (B/Wout) operand strip in-cache.

#define VOCAB 10000
#define VPAD  10240   // VOCAB padded to a multiple of 64 (4 N-tiles per wave)
#define EMB   512
#define HID   1024
#define TSTEP 128
#define BATCH 64

typedef __bf16 bf16;
typedef __attribute__((ext_vector_type(16))) __bf16 v16bf;
typedef __attribute__((ext_vector_type(8)))  __bf16 v8bf;
typedef __attribute__((ext_vector_type(8)))  float  v8f;

// ---- fragment loaders (layouts per CDNA5 ISA 7.12.2) ----
// ubase is wave-uniform and loop-INVARIANT; off is per-lane, carries k.

__device__ __forceinline__ int lane_off_a(int lda, int lane) {
  return (lane & 15) * lda + ((lane & 16) >> 1);   // row*lda + (lanes>=16 ? 8 : 0)
}
__device__ __forceinline__ int lane_off_b(int ldw, int lane) {
  return (lane & 15) * ldw + ((lane & 16) >> 1);   // col*ldw + (lanes>=16 ? 8 : 0)
}

__device__ __forceinline__ v16bf frag_a_u(const bf16* __restrict__ ubase, int off) {
  __builtin_assume(off >= 0);
  v8bf lo = *(const v8bf*)(ubase + off);           // K = kh .. kh+7      (VGPR 0-3)
  v8bf hi = *(const v8bf*)(ubase + off + 16);      // K = kh+16 .. kh+23  (VGPR 4-7)
  v16bf a;
#pragma unroll
  for (int i = 0; i < 8; ++i) { a[i] = lo[i]; a[i + 8] = hi[i]; }
  return a;
}

__device__ __forceinline__ v16bf frag_b_u(const bf16* __restrict__ ubase, int off) {
  __builtin_assume(off >= 0);
  v8bf lo = *(const v8bf*)(ubase + off);           // K = kb .. kb+7
  v8bf hi = *(const v8bf*)(ubase + off + 16);      // K = kb+16 .. kb+23
  v16bf b;
#pragma unroll
  for (int i = 0; i < 8; ++i) { b[i] = lo[i]; b[i + 8] = hi[i]; }
  return b;
}

__device__ __forceinline__ v8f wmma_bf16(v16bf a, v16bf b, v8f c) {
  return __builtin_amdgcn_wmma_f32_16x16x32_bf16(false, a, false, b, (short)0, c, false, false);
}

__device__ __forceinline__ v8f vzero8() {
  v8f z;
#pragma unroll
  for (int i = 0; i < 8; ++i) z[i] = 0.0f;
  return z;
}

// ---- elementwise helpers ----

__global__ __launch_bounds__(256)
void cast_f32_bf16_kernel(const float* __restrict__ src, bf16* __restrict__ dst, int n) {
  for (int i = blockIdx.x * blockDim.x + threadIdx.x; i < n; i += gridDim.x * blockDim.x)
    dst[i] = (bf16)src[i];
}

__global__ __launch_bounds__(256)
void fill0_bf16_kernel(bf16* __restrict__ dst, int n) {
  for (int i = blockIdx.x * blockDim.x + threadIdx.x; i < n; i += gridDim.x * blockDim.x)
    dst[i] = (bf16)0.0f;
}

__global__ __launch_bounds__(256)
void addvec_kernel(const float* __restrict__ a, const float* __restrict__ b,
                   float* __restrict__ o, int n) {
  int i = blockIdx.x * blockDim.x + threadIdx.x;
  if (i < n) o[i] = a[i] + b[i];
}

// X[t*B+b, e] = emb[inputs[t*B+b], e]  (bf16)
__global__ __launch_bounds__(256)
void embed_kernel(const int* __restrict__ inputs, const float* __restrict__ emb,
                  bf16* __restrict__ X) {
  const int total = TSTEP * BATCH * EMB;
  for (int i = blockIdx.x * blockDim.x + threadIdx.x; i < total; i += gridDim.x * blockDim.x) {
    int row = i / EMB;
    int e   = i - row * EMB;
    int tok = inputs[row];
    X[i] = (bf16)emb[(size_t)tok * EMB + e];
  }
}

// ---- batched GEMM: C[M,NC] f32 = A[M,K]bf16 @ W[NW,K]bf16^T + bias ----
// W may be padded: NW >= NC; K-loop runs unguarded over padded tile space,
// only the epilogue store checks colbase < NC (wave-uniform, no WMMA inside).
// One wave computes a (16*MT)x(16*NT) block; B fragments reused across MT.
// M-groups fastest in wid: co-resident waves share the B strip in-cache.
// M % (16*MT) == 0, NW % (16*NT) == 0, K % 64 == 0 (template constants).
template<int MT, int NT, int M, int NW, int K, int NC>
__global__ __launch_bounds__(256)
void gemm_bias_kernel(const bf16* __restrict__ A, const bf16* __restrict__ W,
                      const float* __restrict__ bias, float* __restrict__ C) {
  const int lane = threadIdx.x & 31;
  const int wid  = __builtin_amdgcn_readfirstlane((blockIdx.x * blockDim.x + threadIdx.x) >> 5);
  constexpr int mgroups = M / (16 * MT);
  constexpr int ngroups = NW / (16 * NT);
  if (wid >= mgroups * ngroups) return;   // scalar branch (wid is SGPR)
  const int ng = wid / mgroups;           // M fastest -> block shares B strip
  const int mg = wid - ng * mgroups;

  // wave-uniform, loop-invariant row bases (SGPR pairs)
  const bf16* Ab[MT];
  const bf16* Wb[NT];
#pragma unroll
  for (int i = 0; i < MT; ++i) Ab[i] = A + (size_t)((mg * MT + i) * 16) * K;
#pragma unroll
  for (int j = 0; j < NT; ++j) Wb[j] = W + (size_t)((ng * NT + j) * 16) * K;

  // per-lane 32-bit offsets; k is folded into these (loop-variant VGPR part)
  const int aoff = lane_off_a(K, lane);
  const int boff = lane_off_b(K, lane);

  v8f acc[MT][NT];
#pragma unroll
  for (int i = 0; i < MT; ++i)
#pragma unroll
    for (int j = 0; j < NT; ++j) acc[i][j] = vzero8();

  v16bf a0[MT], b0[NT], a1[MT], b1[NT];

  auto loadset = [&](v16bf (&as)[MT], v16bf (&bs)[NT], int koff) {
    const int ao = aoff + koff;
    const int bo = boff + koff;
#pragma unroll
    for (int i = 0; i < MT; ++i) as[i] = frag_a_u(Ab[i], ao);
#pragma unroll
    for (int j = 0; j < NT; ++j) bs[j] = frag_b_u(Wb[j], bo);
  };
  auto compset = [&](v16bf (&as)[MT], v16bf (&bs)[NT]) {
#pragma unroll
    for (int j = 0; j < NT; ++j)
#pragma unroll
      for (int i = 0; i < MT; ++i) acc[i][j] = wmma_bf16(as[i], bs[j], acc[i][j]);
  };

  // 2-deep software pipeline: K/32 steps (even count since K % 64 == 0)
  loadset(a0, b0, 0);
  int k = 32;
  for (; k < K - 32; k += 64) {
    loadset(a1, b1, k);        // prefetch step s+1
    compset(a0, b0);           // compute step s
    loadset(a0, b0, k + 32);   // prefetch step s+2
    compset(a1, b1);           // compute step s+1
  }
  loadset(a1, b1, K - 32);     // last step's loads
  compset(a0, b0);
  compset(a1, b1);

  const int ncol = lane & 15;
#pragma unroll
  for (int i = 0; i < MT; ++i) {
    const int mbase = (mg * MT + i) * 16 + ((lane & 16) >> 1);  // +8 for lanes 16-31
#pragma unroll
    for (int j = 0; j < NT; ++j) {
      const int colbase = (ng * NT + j) * 16;
      if (colbase < NC) {                      // folds away when NW == NC
        const int col = colbase + ncol;
        const float bv = bias ? bias[col] : 0.0f;
        const int idx0 = mbase * NC + col;     // < 2^31 for all our shapes
        __builtin_assume(idx0 >= 0);
#pragma unroll
        for (int r = 0; r < 8; ++r) C[idx0 + r * NC] = acc[i][j][r] + bv;
      }
    }
  }
}

// ---- layer-0 step: out = tanh(addend + A1 @ W1^T) ----
// M=BATCH=64 (4 mtiles), N=K=HID. 64 waves; 5 streams -> full 2-deep pipeline.
__global__ __launch_bounds__(256)
void rnn_step1_kernel(const bf16* __restrict__ A1, const bf16* __restrict__ W1,
                      const float* __restrict__ addend,
                      bf16* __restrict__ outBf, float* __restrict__ outF32) {
  const int lane = threadIdx.x & 31;
  const int wid  = __builtin_amdgcn_readfirstlane((blockIdx.x * blockDim.x + threadIdx.x) >> 5);
  if (wid >= 64) return;
  const int mtile = wid & 3;              // M fastest: block shares W strip
  const int n0    = (wid >> 2) * 4;

  const bf16* A1r = A1 + (size_t)(mtile * 16) * HID;
  const bf16* W1b[4];
#pragma unroll
  for (int j = 0; j < 4; ++j) W1b[j] = W1 + (size_t)((n0 + j) * 16) * HID;

  const int aoff = lane_off_a(HID, lane);
  const int boff = lane_off_b(HID, lane);

  v8f acc[4] = {vzero8(), vzero8(), vzero8(), vzero8()};
  v16bf ac, an;
  v16bf bc[4], bn[4];

  auto loadset = [&](v16bf& fa, v16bf (&fb)[4], int koff) {
    const int ao = aoff + koff;
    const int bo = boff + koff;
    fa = frag_a_u(A1r, ao);
#pragma unroll
    for (int j = 0; j < 4; ++j) fb[j] = frag_b_u(W1b[j], bo);
  };
  auto compset = [&](v16bf& fa, v16bf (&fb)[4]) {
#pragma unroll
    for (int j = 0; j < 4; ++j) acc[j] = wmma_bf16(fa, fb[j], acc[j]);
  };

  loadset(ac, bc, 0);
  int k = 32;
  for (; k < HID - 32; k += 64) {
    loadset(an, bn, k);
    compset(ac, bc);
    loadset(ac, bc, k + 32);
    compset(an, bn);
  }
  loadset(an, bn, HID - 32);
  compset(ac, bc);
  compset(an, bn);

  const int mbase = mtile * 16 + ((lane & 16) >> 1);
  const int ncol  = lane & 15;
#pragma unroll
  for (int j = 0; j < 4; ++j) {
    const int idx0 = mbase * HID + (n0 + j) * 16 + ncol;
    __builtin_assume(idx0 >= 0);
#pragma unroll
    for (int r = 0; r < 8; ++r) {
      const int idx = idx0 + r * HID;
      float v = tanhf(acc[j][r] + addend[idx]);
      outBf[idx] = (bf16)v;
      if (outF32) outF32[idx] = v;
    }
  }
}

// ---- layer-1 step: out = tanh(A1 @ W1^T + A2 @ W2^T + bias) ----
// 10 streams: no cross-iteration double-buffer (would spill); instead all 10
// fragment loads of a k-step are batched ahead of its 8 WMMAs (staggered waits).
__global__ __launch_bounds__(256)
void rnn_step2_kernel(const bf16* __restrict__ A1, const bf16* __restrict__ W1,
                      const bf16* __restrict__ A2, const bf16* __restrict__ W2,
                      const float* __restrict__ bias,
                      bf16* __restrict__ outBf, float* __restrict__ outF32) {
  const int lane = threadIdx.x & 31;
  const int wid  = __builtin_amdgcn_readfirstlane((blockIdx.x * blockDim.x + threadIdx.x) >> 5);
  if (wid >= 64) return;
  const int mtile = wid & 3;              // M fastest: block shares W strips
  const int n0    = (wid >> 2) * 4;

  const bf16* A1r = A1 + (size_t)(mtile * 16) * HID;
  const bf16* A2r = A2 + (size_t)(mtile * 16) * HID;
  const bf16* W1b[4];
  const bf16* W2b[4];
#pragma unroll
  for (int j = 0; j < 4; ++j) {
    W1b[j] = W1 + (size_t)((n0 + j) * 16) * HID;
    W2b[j] = W2 + (size_t)((n0 + j) * 16) * HID;
  }

  const int aoff = lane_off_a(HID, lane);
  const int boff = lane_off_b(HID, lane);

  v8f acc[4] = {vzero8(), vzero8(), vzero8(), vzero8()};

  for (int k = 0; k < HID; k += 32) {
    const int ao = aoff + k;
    const int bo = boff + k;
    v16bf fa1 = frag_a_u(A1r, ao);
    v16bf fa2 = frag_a_u(A2r, ao);
    v16bf fb1[4], fb2[4];
#pragma unroll
    for (int j = 0; j < 4; ++j) fb1[j] = frag_b_u(W1b[j], bo);
#pragma unroll
    for (int j = 0; j < 4; ++j) fb2[j] = frag_b_u(W2b[j], bo);
#pragma unroll
    for (int j = 0; j < 4; ++j) acc[j] = wmma_bf16(fa1, fb1[j], acc[j]);
#pragma unroll
    for (int j = 0; j < 4; ++j) acc[j] = wmma_bf16(fa2, fb2[j], acc[j]);
  }

  const int mbase = mtile * 16 + ((lane & 16) >> 1);
  const int ncol  = lane & 15;
#pragma unroll
  for (int j = 0; j < 4; ++j) {
    const int col  = (n0 + j) * 16 + ncol;
    const float bv = bias[col];
    const int idx0 = mbase * HID + col;
    __builtin_assume(idx0 >= 0);
#pragma unroll
    for (int r = 0; r < 8; ++r) {
      const int idx = idx0 + r * HID;
      float v = tanhf(acc[j][r] + bv);
      outBf[idx] = (bf16)v;
      if (outF32) outF32[idx] = v;
    }
  }
}

// ---- host ----
static inline int ceil_div_i(int a, int b) { return (a + b - 1) / b; }

extern "C" void kernel_launch(void* const* d_in, const int* in_sizes, int n_in,
                              void* d_out, int out_size, void* d_ws, size_t ws_size,
                              hipStream_t stream) {
  (void)in_sizes; (void)n_in; (void)out_size; (void)ws_size;
  const int*   inputs = (const int*)  d_in[0];
  const float* hidden = (const float*)d_in[1];
  const float* emb    = (const float*)d_in[2];
  const float* Wx0    = (const float*)d_in[3];
  const float* bx0    = (const float*)d_in[4];
  const float* Wh0    = (const float*)d_in[5];
  const float* bh0    = (const float*)d_in[6];
  const float* Wx1    = (const float*)d_in[7];
  const float* bx1    = (const float*)d_in[8];
  const float* Wh1    = (const float*)d_in[9];
  const float* bh1    = (const float*)d_in[10];
  const float* Wout   = (const float*)d_in[11];
  const float* bout   = (const float*)d_in[12];

  float* logits  = (float*)d_out;                               // [T*B, VOCAB]
  float* hid_out = logits + (size_t)TSTEP * BATCH * VOCAB;      // [2, B, HID]

  // workspace carve (256B aligned)
  unsigned char* wsp = (unsigned char*)d_ws;
  auto take = [&](size_t bytes) -> void* {
    void* p = (void*)wsp;
    wsp += (bytes + 255) & ~(size_t)255;
    return p;
  };
  bf16* Xbf   = (bf16*)take((size_t)TSTEP * BATCH * EMB * 2);
  bf16* Wx0b  = (bf16*)take((size_t)HID * EMB * 2);
  bf16* Wh0b  = (bf16*)take((size_t)HID * HID * 2);
  bf16* Wx1b  = (bf16*)take((size_t)HID * HID * 2);
  bf16* Wh1b  = (bf16*)take((size_t)HID * HID * 2);
  bf16* Woutb = (bf16*)take((size_t)VPAD * HID * 2);            // padded to 10240 rows
  float* Xproj = (float*)take((size_t)TSTEP * BATCH * HID * 4); // includes bx0+bh0
  bf16* H0    = (bf16*)take((size_t)(TSTEP + 1) * BATCH * HID * 2);
  bf16* H1    = (bf16*)take((size_t)(TSTEP + 1) * BATCH * HID * 2);
  float* bsum0 = (float*)take((size_t)HID * 4);
  float* bsum1 = (float*)take((size_t)HID * 4);

  // 1) casts + embedding (+ zero-fill Wout pad rows)
  cast_f32_bf16_kernel<<<256,  256, 0, stream>>>(Wx0, Wx0b, HID * EMB);
  cast_f32_bf16_kernel<<<512,  256, 0, stream>>>(Wh0, Wh0b, HID * HID);
  cast_f32_bf16_kernel<<<512,  256, 0, stream>>>(Wx1, Wx1b, HID * HID);
  cast_f32_bf16_kernel<<<512,  256, 0, stream>>>(Wh1, Wh1b, HID * HID);
  cast_f32_bf16_kernel<<<4096, 256, 0, stream>>>(Wout, Woutb, VOCAB * HID);
  fill0_bf16_kernel<<<240, 256, 0, stream>>>(Woutb + (size_t)VOCAB * HID,
                                             (VPAD - VOCAB) * HID);
  cast_f32_bf16_kernel<<<64,   256, 0, stream>>>(hidden,             H0, BATCH * HID); // h0(-1)
  cast_f32_bf16_kernel<<<64,   256, 0, stream>>>(hidden + BATCH*HID, H1, BATCH * HID); // h1(-1)
  embed_kernel<<<4096, 256, 0, stream>>>(inputs, emb, Xbf);
  addvec_kernel<<<ceil_div_i(HID, 256), 256, 0, stream>>>(bx0, bh0, bsum0, HID);
  addvec_kernel<<<ceil_div_i(HID, 256), 256, 0, stream>>>(bx1, bh1, bsum1, HID);

  // 2) Xproj[T*B, HID] = Xbf @ Wx0^T + (bx0+bh0)   -- <2,4>
  {
    constexpr int waves = (TSTEP * BATCH / 32) * (HID / 64);     // 256 * 16 = 4096
    gemm_bias_kernel<2, 4, TSTEP * BATCH, HID, EMB, HID>
        <<<ceil_div_i(waves, 8), 256, 0, stream>>>(Xbf, Wx0b, bsum0, Xproj);
  }

  // 3) sequential recurrence: 2 launches per step
  for (int t = 0; t < TSTEP; ++t) {
    const size_t off = (size_t)t * BATCH * HID;
    float* f0 = (t == TSTEP - 1) ? hid_out : nullptr;
    float* f1 = (t == TSTEP - 1) ? hid_out + (size_t)BATCH * HID : nullptr;
    // h0(t) = tanh(Xproj[t] + h0(t-1) @ Wh0^T)       (biases folded into Xproj)
    rnn_step1_kernel<<<8, 256, 0, stream>>>(H0 + off, Wh0b, Xproj + off,
                                            H0 + off + (size_t)BATCH * HID, f0);
    // h1(t) = tanh(h0(t) @ Wx1^T + h1(t-1) @ Wh1^T + (bx1+bh1))
    rnn_step2_kernel<<<8, 256, 0, stream>>>(H0 + off + (size_t)BATCH * HID, Wx1b,
                                            H1 + off, Wh1b, bsum1,
                                            H1 + off + (size_t)BATCH * HID, f1);
  }

  // 4) logits[T*B, VOCAB] = H1[1..T] @ Wout^T + bout -- <2,4> over padded N
  {
    constexpr int waves = (TSTEP * BATCH / 32) * (VPAD / 64);    // 256 * 160 = 40960
    gemm_bias_kernel<2, 4, TSTEP * BATCH, VPAD, HID, VOCAB>
        <<<ceil_div_i(waves, 8), 256, 0, stream>>>(H1 + (size_t)BATCH * HID, Woutb, bout, logits);
  }
}
